// RayRenderer_63694365000008
// MI455X (gfx1250) — matrix operand order repaired and checked by reading, not compiled, not run on hardware
//
#include <hip/hip_runtime.h>
#include <math.h>

typedef __attribute__((ext_vector_type(16))) _Float16 v16h;
typedef __attribute__((ext_vector_type(8)))  float    v8f;

#define NERF_N_RAYS    8192
#define NERF_N_SAMPLES 64
#define NERF_NPTS      (NERF_N_RAYS * NERF_N_SAMPLES)

union AFrag {
    v16h v;
    uint4 q[2];
};

struct MlpArgs {
    const float* rays_o;
    const float* rays_d;
    const int*   nearp;     // scalar, int32-or-float32 bits
    const int*   farp;
    const _Float16* wpk[9]; // packed f16 B-fragment weights: 8 hidden + feat
    const float*    bias[9];
    const float* dens_W;    // 128
    const float* dens_b;    // 1
    const float* rgb_W;     // 152 x 3 row-major
    const float* rgb_b;     // 3
    float4* pts;            // per-point {r,g,b,sigma}
};

// Robustly interpret a 4-byte scalar that may be int32 (python int) or float32.
__device__ __forceinline__ float scalar_to_float(const int* p) {
    int iv = *p;
    float fv = __int_as_float(iv);
    float a = fabsf(fv);
    if (a >= 1e-30f && a <= 1e30f) return fv;   // plausible float bit pattern
    return (float)iv;                           // small int bit pattern
}

// A-fragment (16x32 f16, M x K) from an LDS row-major buffer (stride 256).
// Lane l<16: row=l, K = k0+0..7 and k0+16..23; lane l>=16: row=l-16, K = k0+8..15, k0+24..31.
__device__ __forceinline__ v16h load_a_frag(const _Float16* src, int k0, int lane) {
    int row = lane & 15;
    int kb  = k0 + ((lane >> 4) << 3);
    const _Float16* p = src + row * 256 + kb;
    AFrag u;
    u.q[0] = *(const uint4*)(p);        // K offsets +0..7
    u.q[1] = *(const uint4*)(p + 16);   // K offsets +16..23
    return u.v;
}

// B-fragment (32x16 f16, K x N) from the prepacked global weight buffer.
// Tile = 32 lanes x 16 contiguous halves per lane (1KB).
__device__ __forceinline__ v16h load_b_frag(const _Float16* wtile, int lane) {
    const _Float16* p = wtile + lane * 16;
    AFrag u;
    u.q[0] = *(const uint4*)(p);
    u.q[1] = *(const uint4*)(p + 8);
    return u.v;
}

// Hoist a full layer input (16 x 256) from the wave's LDS buffer into registers.
__device__ __forceinline__ void load_act(v16h* act, const _Float16* buf, int lane) {
#pragma unroll
    for (int kt = 0; kt < 8; ++kt) act[kt] = load_a_frag(buf, kt * 32, lane);
}

// Positional-encoding value for column c (63 real cols + 1 zero pad).
__device__ __forceinline__ float posenc_val(int c, float px, float py, float pz) {
    if (c >= 63) return 0.0f;
    if (c < 3)  return (c == 0) ? px : ((c == 1) ? py : pz);
    int idx = c - 3;
    int f   = idx / 6;
    int j   = idx - f * 6;
    int ax  = (j >= 3) ? (j - 3) : j;
    float pc = (ax == 0) ? px : ((ax == 1) ? py : pz);
    float x  = pc * (float)(1 << f);
    return (j < 3) ? sinf(x) : cosf(x);
}

// One dense layer with register-resident A and 4-way N-tile interleaving:
//   buf[16 x (NTILES*16)] = act( A @ W + b ),   A = aMain(KT frags) ++ aSkip(KT2 frags)
// Four independent accumulators are interleaved so a WMMA's same-accumulator
// successor is 4 issue slots away (fills the f16 WMMA->WMMA 4-coexec hazard
// window with useful WMMAs instead of v_nops).
// Output written back into the SAME per-wave LDS buffer (A lives in registers).
template<int KT, int KT2, int NTILES, bool RELU>
__device__ __forceinline__ void dense_layer(
    const v16h* aMain, const v16h* aSkip,
    const _Float16* __restrict__ wpk, const float* __restrict__ bias,
    _Float16* dstBuf, int lane)
{
    constexpr int KTT = KT + KT2;
    static_assert((NTILES & 3) == 0, "NTILES must be a multiple of 4");
    int nsub = lane & 15;
    int mo   = (lane >> 4) << 3;
#pragma unroll 1
    for (int nt = 0; nt < NTILES; nt += 4) {
        const _Float16* wt[4];
        v8f c[4];
#pragma unroll
        for (int u = 0; u < 4; ++u) {
            wt[u] = wpk + (size_t)(nt + u) * (size_t)(KTT * 512);
            float bv = bias[(nt + u) * 16 + nsub];
#pragma unroll
            for (int r = 0; r < 8; ++r) c[u][r] = bv;
        }
        __builtin_prefetch(wt[3] + KTT * 512, 0, 1);   // next group's weights -> WGP$

#pragma unroll
        for (int kt = 0; kt < KT; ++kt) {
#pragma unroll
            for (int u = 0; u < 4; ++u) {
                v16h b = load_b_frag(wt[u] + kt * 512, lane);
                c[u] = __builtin_amdgcn_wmma_f32_16x16x32_f16(
                           false, aMain[kt], false, b, (short)0, c[u], false, false);
            }
        }
#pragma unroll
        for (int kt = 0; kt < KT2; ++kt) {
#pragma unroll
            for (int u = 0; u < 4; ++u) {
                v16h b = load_b_frag(wt[u] + (KT + kt) * 512, lane);
                c[u] = __builtin_amdgcn_wmma_f32_16x16x32_f16(
                           false, aSkip[kt], false, b, (short)0, c[u], false, false);
            }
        }

        // C layout: VGPR r, lanes<16 -> M=r, lanes>=16 -> M=r+8; N = lane&15.
#pragma unroll
        for (int u = 0; u < 4; ++u) {
            int n = (nt + u) * 16 + nsub;
#pragma unroll
            for (int r = 0; r < 8; ++r) {
                float f = c[u][r];
                if (RELU) f = fmaxf(f, 0.0f);
                dstBuf[(mo + r) * 256 + n] = (_Float16)f;
            }
        }
    }
}

// ---------------------------------------------------------------------------
// Prepack: f32 (K x N row-major) -> f16 B-fragment tiles, K zero-padded.
// ---------------------------------------------------------------------------
__global__ void nerf_prepack_kernel(const float* __restrict__ W, int K, int Kpad, int N,
                                    _Float16* __restrict__ dst) {
    int e = blockIdx.x * blockDim.x + threadIdx.x;
    int ktilesL = Kpad >> 5;
    int total = (N >> 4) * ktilesL * 512;
    if (e >= total) return;
    int tile = e >> 9;
    int w    = e & 511;
    int lane = w >> 4;
    int j    = w & 15;
    int kt = tile % ktilesL;
    int nt = tile / ktilesL;
    int k = kt * 32 + ((lane >> 4) << 3) + ((j < 8) ? j : (8 + j));
    int n = nt * 16 + (lane & 15);
    float v = (k < K) ? W[k * N + n] : 0.0f;
    dst[e] = (_Float16)v;
}

// ---------------------------------------------------------------------------
// Main MLP kernel: 4 wave32 per WG, each wave independently owns 16 points.
// No barriers; activations live in registers between layers; one 8KB LDS
// buffer per wave, overwritten in place.
// ---------------------------------------------------------------------------
__global__ __launch_bounds__(128) void nerf_mlp_kernel(MlpArgs args) {
    __shared__ _Float16 lbuf[4][16 * 256];   // 32 KB

    int tid  = threadIdx.x;
    int lane = tid & 31;
    int wave = tid >> 5;
    _Float16* buf = lbuf[wave];
    int mbase = blockIdx.x * 64 + wave * 16;

    float nearv = scalar_to_float(args.nearp);
    float farv  = scalar_to_float(args.farp);

    // --- per-lane point for this lane's A-fragment row ---
    int row = lane & 15;
    int g   = mbase + row;
    int ray = g >> 6;                 // N_SAMPLES == 64
    int s   = g & 63;
    float t  = nearv + (farv - nearv) * ((float)s * (1.0f / 63.0f));
    float px = args.rays_o[ray * 3 + 0] + t * args.rays_d[ray * 3 + 0];
    float py = args.rays_o[ray * 3 + 1] + t * args.rays_d[ray * 3 + 1];
    float pz = args.rays_o[ray * 3 + 2] + t * args.rays_d[ray * 3 + 2];

    // --- positional encoding directly in WMMA A-fragment layout (2 frags) ---
    v16h pfrag[2];
#pragma unroll
    for (int kt = 0; kt < 2; ++kt) {
        int kb = kt * 32 + ((lane >> 4) << 3);
        v16h r;
#pragma unroll
        for (int e = 0; e < 16; ++e) {
            int c = kb + ((e < 8) ? e : (8 + e));   // e>=8 -> K offset +16
            r[e] = (_Float16)posenc_val(c, px, py, pz);
        }
        pfrag[kt] = r;
    }

    v16h act[8];
    // L0: posenc(63->64) -> 256
    dense_layer<2, 0, 16, true >(pfrag, pfrag, args.wpk[0], args.bias[0], buf, lane);
    load_act(act, buf, lane);
    dense_layer<8, 0, 16, true >(act, pfrag, args.wpk[1], args.bias[1], buf, lane);
    load_act(act, buf, lane);
    dense_layer<8, 0, 16, true >(act, pfrag, args.wpk[2], args.bias[2], buf, lane);
    load_act(act, buf, lane);
    dense_layer<8, 0, 16, true >(act, pfrag, args.wpk[3], args.bias[3], buf, lane);
    load_act(act, buf, lane);
    dense_layer<8, 0, 16, true >(act, pfrag, args.wpk[4], args.bias[4], buf, lane);
    load_act(act, buf, lane);
    // L5: input = concat(h(256), posenc(64)) -> 256   (skip connection from regs)
    dense_layer<8, 2, 16, true >(act, pfrag, args.wpk[5], args.bias[5], buf, lane);
    load_act(act, buf, lane);
    dense_layer<8, 0, 16, true >(act, pfrag, args.wpk[6], args.bias[6], buf, lane);
    load_act(act, buf, lane);
    dense_layer<8, 0, 16, true >(act, pfrag, args.wpk[7], args.bias[7], buf, lane);
    load_act(act, buf, lane);
    // feature head: 256 -> 128, no ReLU (cols 0..127 of buf)
    dense_layer<8, 0, 8, false>(act, pfrag, args.wpk[8], args.bias[8], buf, lane);

    // --- density + rgb heads (tiny): one lane per point within the wave ---
    if (lane < 16) {
        int p = lane;
        int gg = mbase + p;
        int rr = gg >> 6;
        const _Float16* feat = buf + p * 256;

        float d  = args.dens_b[0];
        float r0 = args.rgb_b[0], r1 = args.rgb_b[1], r2 = args.rgb_b[2];
        for (int k = 0; k < 128; ++k) {
            float fv = (float)feat[k];
            d  += fv * args.dens_W[k];
            r0 += fv * args.rgb_W[k * 3 + 0];
            r1 += fv * args.rgb_W[k * 3 + 1];
            r2 += fv * args.rgb_W[k * 3 + 2];
        }
        d = fmaxf(d, 0.0f);

        // view encoding: 4 freqs x [sin(xyz), cos(xyz)] = 24 extra rgb inputs
        for (int f = 0; f < 4; ++f) {
            float sc = (float)(1 << f);
            for (int j = 0; j < 6; ++j) {
                int ax = j % 3;
                float x  = args.rays_d[rr * 3 + ax] * sc;
                float ve = (j < 3) ? sinf(x) : cosf(x);
                int col = 128 + f * 6 + j;
                r0 += ve * args.rgb_W[col * 3 + 0];
                r1 += ve * args.rgb_W[col * 3 + 1];
                r2 += ve * args.rgb_W[col * 3 + 2];
            }
        }
        float4 o;
        o.x = 1.0f / (1.0f + expf(-r0));
        o.y = 1.0f / (1.0f + expf(-r1));
        o.z = 1.0f / (1.0f + expf(-r2));
        o.w = d;
        args.pts[gg] = o;
    }
}

// ---------------------------------------------------------------------------
// Alpha compositing: one thread per ray, sequential over 64 samples.
// ---------------------------------------------------------------------------
__global__ void nerf_composite_kernel(const float4* __restrict__ pts,
                                      const int* nearp, const int* farp,
                                      float* __restrict__ out) {
    int ray = blockIdx.x * blockDim.x + threadIdx.x;
    if (ray >= NERF_N_RAYS) return;
    float nearv = scalar_to_float(nearp);
    float farv  = scalar_to_float(farp);
    float dt = (farv - nearv) * (1.0f / 63.0f);
    float T = 1.0f;
    float r = 0.0f, g = 0.0f, b = 0.0f;
    for (int s = 0; s < NERF_N_SAMPLES; ++s) {
        float4 pt = pts[ray * NERF_N_SAMPLES + s];
        float dist  = (s == NERF_N_SAMPLES - 1) ? 1e10f : dt;
        float alpha = 1.0f - expf(-pt.w * dist);
        float w = alpha * T;
        r += w * pt.x;
        g += w * pt.y;
        b += w * pt.z;
        T *= (1.0f - alpha);
    }
    out[ray * 3 + 0] = r;
    out[ray * 3 + 1] = g;
    out[ray * 3 + 2] = b;
}

extern "C" void kernel_launch(void* const* d_in, const int* in_sizes, int n_in,
                              void* d_out, int out_size, void* d_ws, size_t ws_size,
                              hipStream_t stream) {
    (void)out_size; (void)ws_size;
    // Top-level dict order: rays_o, rays_d, near, far, <params...>
    const float* rays_o = (const float*)d_in[0];
    const float* rays_d = (const float*)d_in[1];
    const int*   nearp  = (const int*)d_in[2];
    const int*   farp   = (const int*)d_in[3];

    // Identify params by size pattern (robust to dict-order vs sorted flatten).
    int w0 = -1, b0 = -1, fw = -1, i456 = -1, i3 = -1, i1 = -1;
    int idx128[2] = {-1, -1}; int n128 = 0;
    for (int i = 4; i < n_in; ++i) {
        int s = in_sizes[i];
        if (s == 16128 && w0 < 0) w0 = i;                 // hidden_W[0] (63x256)
        else if (s == 32768 && fw < 0) fw = i;            // feat_W (256x128)
        else if (s == 128 && n128 < 2) idx128[n128++] = i;// feat_b / dens_W
        else if (s == 456 && i456 < 0) i456 = i;          // rgb_W
        else if (s == 3 && i3 < 0) i3 = i;                // rgb_b
        else if (s == 1 && i1 < 0) i1 = i;                // dens_b
        else if (s == 256 && b0 < 0) {                    // 8 consecutive biases
            bool ok = (i + 7 < n_in);
            for (int j = 0; ok && j < 8; ++j) ok = (in_sizes[i + j] == 256);
            if (ok) b0 = i;
        }
    }
    if (w0 < 0 || b0 < 0 || fw < 0 || n128 < 2 || i456 < 0 || i3 < 0 || i1 < 0) return;

    const float* hW[8]; const float* hb[8];
    for (int j = 0; j < 8; ++j) hW[j] = (const float*)d_in[w0 + j]; // list is contiguous
    for (int j = 0; j < 8; ++j) hb[j] = (const float*)d_in[b0 + j];
    const float* featW = (const float*)d_in[fw];
    const float* featb; const float* densW;
    if (idx128[0] < fw) { densW = (const float*)d_in[idx128[0]]; featb = (const float*)d_in[idx128[1]]; }
    else                { featb = (const float*)d_in[idx128[0]]; densW = (const float*)d_in[idx128[1]]; }
    const float* densb = (const float*)d_in[i1];
    const float* rgbW  = (const float*)d_in[i456];
    const float* rgbb  = (const float*)d_in[i3];

    // Workspace layout: [packed f16 weights (1MB)] [float4 per-point buffer (8MB)]
    static const int Kh[9]   = {63, 256, 256, 256, 256, 319, 256, 256, 256};
    static const int Kpad[9] = {64, 256, 256, 256, 256, 320, 256, 256, 256};
    static const int Nn[9]   = {256, 256, 256, 256, 256, 256, 256, 256, 128};
    const float* Wsrc[9] = {hW[0], hW[1], hW[2], hW[3], hW[4], hW[5], hW[6], hW[7], featW};
    const float* Bsrc[9] = {hb[0], hb[1], hb[2], hb[3], hb[4], hb[5], hb[6], hb[7], featb};

    _Float16* pkBase = (_Float16*)d_ws;
    size_t off = 0;
    _Float16* pk[9];
    for (int l = 0; l < 9; ++l) {
        pk[l] = pkBase + off;
        off += (size_t)(Nn[l] / 16) * (size_t)(Kpad[l] / 32) * 512;
    }
    size_t pkBytes = (off * sizeof(_Float16) + 255) & ~(size_t)255;
    float4* pts = (float4*)((char*)d_ws + pkBytes);

    // 1) prepack weights to f16 B-fragment layout
    for (int l = 0; l < 9; ++l) {
        int total = (Nn[l] / 16) * (Kpad[l] / 32) * 512;
        nerf_prepack_kernel<<<(total + 255) / 256, 256, 0, stream>>>(
            Wsrc[l], Kh[l], Kpad[l], Nn[l], pk[l]);
    }

    // 2) MLP: 64 points per WG (4 independent waves x 16 points), no barriers
    MlpArgs args;
    args.rays_o = rays_o; args.rays_d = rays_d;
    args.nearp = nearp;   args.farp = farp;
    for (int l = 0; l < 9; ++l) { args.wpk[l] = pk[l]; args.bias[l] = Bsrc[l]; }
    args.dens_W = densW; args.dens_b = densb;
    args.rgb_W = rgbW;   args.rgb_b = rgbb;
    args.pts = pts;
    nerf_mlp_kernel<<<NERF_NPTS / 64, 128, 0, stream>>>(args);

    // 3) per-ray alpha compositing
    nerf_composite_kernel<<<(NERF_N_RAYS + 255) / 256, 256, 0, stream>>>(
        pts, nearp, farp, (float*)d_out);
}
